// Attention_12352325943580
// MI455X (gfx1250) — matrix operand order repaired and checked by reading, not compiled, not run on hardware
//
#include <hip/hip_runtime.h>
#include <hip/hip_bf16.h>

// Problem constants (match reference)
#define E_    512
#define L_    4
#define B_    32
#define SE_   48
#define SD_   48
#define DV_   16000
#define G4E   2048      // 4*E
#define KPAD  576       // (SE + E) padded to multiple of 32

typedef __attribute__((ext_vector_type(16))) __bf16 v16bf;
typedef __attribute__((ext_vector_type(8)))  float  v8f;

union FragU { v16bf v; uint4 q[2]; };

__device__ __forceinline__ v8f wmma_bf16(v16bf a, v16bf b, v8f c) {
  // D = A(16x32 bf16) * B(32x16 bf16) + C(16x16 f32)
  return __builtin_amdgcn_wmma_f32_16x16x32_bf16(
      /*neg_a=*/false, a, /*neg_b=*/false, b,
      /*c_mod=*/(short)0, c, /*reuse_a=*/false, /*reuse_b=*/false);
}

// A fragment: row-major A [M, lda] bf16. 16x32 tile at (m_base, k_base).
// ISA layout: lane m = lane&15; lanes<16: K = {0..7,16..23}; lanes>=16: +8.
__device__ __forceinline__ v16bf load_a_frag(const __bf16* __restrict__ A, int lda,
                                             int m_base, int k_base, int lane) {
  int m = m_base + (lane & 15);
  int khalf = (lane >> 4) & 1;
  const __bf16* p = A + (size_t)m * lda + k_base + khalf * 8;
  FragU f;
  f.q[0] = *(const uint4*)(p);        // K offs +0..7
  f.q[1] = *(const uint4*)(p + 16);   // K offs +16..23
  return f.v;
}

// B fragment: B = W^T where W row-major [N, ldb]; B[k][n] = W[n][k].
// ISA layout: lane n = lane&15; lanes<16: K = {0..7,16..23}; lanes>=16: +8.
__device__ __forceinline__ v16bf load_b_frag(const __bf16* __restrict__ W, int ldb,
                                             int n_base, int k_base, int lane) {
  int n = n_base + (lane & 15);
  int koff = ((lane >> 4) & 1) << 3;
  const __bf16* p = W + (size_t)n * ldb + k_base + koff;
  FragU f;
  f.q[0] = *(const uint4*)(p);        // K offs +0..7
  f.q[1] = *(const uint4*)(p + 16);   // K offs +16..23
  return f.v;
}

// ---------------- prep kernels ----------------

__global__ __launch_bounds__(256) void cvt_bf16_kernel(const float* __restrict__ src,
                                                       __bf16* __restrict__ dst, int n) {
  int i = blockIdx.x * 256 + threadIdx.x;
  if (i < n) dst[i] = (__bf16)src[i];
}

__global__ __launch_bounds__(256) void cvt_w2_pad_kernel(const float* __restrict__ W2,
                                                         __bf16* __restrict__ dst) {
  int i = blockIdx.x * 256 + threadIdx.x;           // over 512*KPAD
  if (i >= E_ * KPAD) return;
  int r = i / KPAD, col = i % KPAD;
  dst[i] = (col < (SE_ + E_)) ? (__bf16)W2[(size_t)r * (SE_ + E_) + col] : (__bf16)0.0f;
}

__global__ __launch_bounds__(256) void bias_sum_kernel(const float* __restrict__ a,
                                                       const float* __restrict__ b,
                                                       float* __restrict__ out, int n) {
  int i = blockIdx.x * 256 + threadIdx.x;
  if (i < n) out[i] = a[i] + b[i];
}

__global__ __launch_bounds__(256) void zero_f32_kernel(float* __restrict__ p, int n) {
  int i = blockIdx.x * 256 + threadIdx.x;
  if (i < n) p[i] = 0.0f;
}

__global__ __launch_bounds__(256) void zero_bf16_kernel(__bf16* __restrict__ p, int n) {
  int i = blockIdx.x * 256 + threadIdx.x;
  if (i < n) p[i] = (__bf16)0.0f;
}

// ---------------- WMMA GEMM kernels ----------------

// gates[B,4E] = Ax[B,E] @ Wih^T + Ah[B,E] @ Whh^T + bias   (all K = 512)
__global__ __launch_bounds__(256)
void lstm_gates_kernel(const __bf16* __restrict__ Ax, const __bf16* __restrict__ Ah,
                       const __bf16* __restrict__ Wih, const __bf16* __restrict__ Whh,
                       const float* __restrict__ bias, float* __restrict__ gates) {
  int wave = threadIdx.x >> 5;
  int lane = threadIdx.x & 31;
  int n_base = (blockIdx.x * 8 + wave) * 16;        // 128 n-tiles over 2048
  v8f acc0 = {}; v8f acc1 = {};
  for (int k = 0; k < E_; k += 32) {
    v16bf bih = load_b_frag(Wih, E_, n_base, k, lane);
    v16bf ax0 = load_a_frag(Ax, E_, 0, k, lane);
    v16bf ax1 = load_a_frag(Ax, E_, 16, k, lane);
    acc0 = wmma_bf16(ax0, bih, acc0);
    acc1 = wmma_bf16(ax1, bih, acc1);
    v16bf bhh = load_b_frag(Whh, E_, n_base, k, lane);
    v16bf ah0 = load_a_frag(Ah, E_, 0, k, lane);
    v16bf ah1 = load_a_frag(Ah, E_, 16, k, lane);
    acc0 = wmma_bf16(ah0, bhh, acc0);
    acc1 = wmma_bf16(ah1, bhh, acc1);
  }
  int n = n_base + (lane & 15);
  int moff = ((lane >> 4) & 1) * 8;
  float bv = bias[n];
#pragma unroll
  for (int v = 0; v < 8; ++v) {
    gates[(size_t)(moff + v) * G4E + n]      = acc0[v] + bv;
    gates[(size_t)(16 + moff + v) * G4E + n] = acc1[v] + bv;
  }
}

// out[B,N] = A[B,K] @ W^T + bias ; W row-major [N,K]; K % 32 == 0
template <bool OUT_BF16>
__global__ __launch_bounds__(256)
void wmma_gemm_kernel(const __bf16* __restrict__ A, const __bf16* __restrict__ W,
                      const float* __restrict__ bias, void* __restrict__ outp,
                      int N, int K) {
  int wave = threadIdx.x >> 5;
  int lane = threadIdx.x & 31;
  int n_base = (blockIdx.x * 8 + wave) * 16;
  if (n_base >= N) return;                          // wave-uniform: EXEC stays all-ones
  v8f acc0 = {}; v8f acc1 = {};
  for (int k = 0; k < K; k += 32) {
    v16bf bb = load_b_frag(W, K, n_base, k, lane);
    v16bf a0 = load_a_frag(A, K, 0, k, lane);
    v16bf a1 = load_a_frag(A, K, 16, k, lane);
    acc0 = wmma_bf16(a0, bb, acc0);
    acc1 = wmma_bf16(a1, bb, acc1);
  }
  int n = n_base + (lane & 15);
  int moff = ((lane >> 4) & 1) * 8;
  float bv = bias[n];
  if (OUT_BF16) {
    __bf16* out = (__bf16*)outp;
#pragma unroll
    for (int v = 0; v < 8; ++v) {
      out[(size_t)(moff + v) * N + n]      = (__bf16)(acc0[v] + bv);
      out[(size_t)(16 + moff + v) * N + n] = (__bf16)(acc1[v] + bv);
    }
  } else {
    float* out = (float*)outp;
#pragma unroll
    for (int v = 0; v < 8; ++v) {
      out[(size_t)(moff + v) * N + n]      = acc0[v] + bv;
      out[(size_t)(16 + moff + v) * N + n] = acc1[v] + bv;
    }
  }
}

// ---------------- LSTM pointwise cell ----------------

__global__ __launch_bounds__(256)
void lstm_cell_kernel(const float* __restrict__ gates, float* __restrict__ h,
                      float* __restrict__ c, __bf16* __restrict__ hbf,
                      const int* __restrict__ e_lengths, int t, int masked,
                      float* __restrict__ upo, int write_out) {
  int idx = blockIdx.x * 256 + threadIdx.x;         // B*E = 16384
  int b = idx >> 9, e = idx & (E_ - 1);
  const float* g = gates + (size_t)b * G4E;
  float gi = g[e], gf = g[E_ + e], gg = g[2 * E_ + e], go = g[3 * E_ + e];
  float si = 1.0f / (1.0f + expf(-gi));
  float sf = 1.0f / (1.0f + expf(-gf));
  float so = 1.0f / (1.0f + expf(-go));
  float nc = sf * c[idx] + si * tanhf(gg);
  float nh = so * tanhf(nc);
  bool keep = masked ? (t < e_lengths[b]) : true;
  float hf = keep ? nh : h[idx];
  float cf = keep ? nc : c[idx];
  h[idx] = hf;
  c[idx] = cf;
  hbf[idx] = (__bf16)hf;
  if (write_out)
    upo[((size_t)b * SE_ + t) * E_ + e] = keep ? nh : 0.0f;
}

// ---------------- gathers ----------------

__global__ __launch_bounds__(256)
void gather_emb1_kernel(const int* __restrict__ e_tokens, const float* __restrict__ emb1,
                        int t, __bf16* __restrict__ x0) {
  int idx = blockIdx.x * 256 + threadIdx.x;         // B*E
  int b = idx >> 9, e = idx & (E_ - 1);
  int tok = e_tokens[b * SE_ + t];
  x0[idx] = (__bf16)emb1[(size_t)tok * E_ + e];
}

__global__ __launch_bounds__(256)
void build_de_in_kernel(const float* __restrict__ ctx, const float* __restrict__ emb2,
                        const int* __restrict__ d_tokens, int t,
                        __bf16* __restrict__ de_in) {
  int idx = blockIdx.x * 256 + threadIdx.x;         // B*KPAD = 18432
  if (idx >= B_ * KPAD) return;
  int b = idx / KPAD, col = idx % KPAD;
  float v;
  if (col < SE_) {
    v = ctx[b * SE_ + col];
  } else if (col < SE_ + E_) {
    int tok = d_tokens[b * SD_ + t];
    v = emb2[(size_t)tok * E_ + (col - SE_)];
  } else {
    v = 0.0f;
  }
  de_in[idx] = (__bf16)v;
}

// ---------------- attention ----------------

__global__ __launch_bounds__(256)
void upo_sum_kernel(const float* __restrict__ upo, float* __restrict__ upo_sum) {
  int idx = blockIdx.x * 256 + threadIdx.x;         // B*SE = 1536
  if (idx >= B_ * SE_) return;
  const float* row = upo + (size_t)idx * E_;
  float s = 0.0f;
  for (int e = 0; e < E_; ++e) s += row[e];
  upo_sum[idx] = s;
}

__global__ __launch_bounds__(256)
void att_dots_kernel(const float* __restrict__ h_top, const float* __restrict__ upo,
                     float* __restrict__ att_raw) {
  int gtid = blockIdx.x * 256 + threadIdx.x;
  int wave = gtid >> 5;                             // one wave per (b,s); 1536 waves
  int lane = gtid & 31;
  int b = wave / SE_, s = wave % SE_;
  const float* hv = h_top + (size_t)b * E_;
  const float* uv = upo + ((size_t)b * SE_ + s) * E_;
  float acc = 0.0f;
  for (int e = lane; e < E_; e += 32) acc += hv[e] * uv[e];
#pragma unroll
  for (int m = 16; m >= 1; m >>= 1) acc += __shfl_xor(acc, m, 32);
  if (lane == 0) att_raw[b * SE_ + s] = acc;
}

__global__ __launch_bounds__(64)
void att_softmax_ctx_kernel(const float* __restrict__ att_raw, const float* __restrict__ W1,
                            const float* __restrict__ b1, const float* __restrict__ upo_sum,
                            float* __restrict__ ctx) {
  __shared__ float sm[SE_];
  __shared__ float red[2];
  int b = blockIdx.x, tid = threadIdx.x;
  if (tid < SE_) {
    float acc = b1[tid];
    const float* ar = att_raw + b * SE_;
    for (int j = 0; j < SE_; ++j) acc += ar[j] * W1[tid * SE_ + j];
    sm[tid] = acc;
  }
  __syncthreads();
  if (tid == 0) {
    float mx = sm[0];
    for (int j = 1; j < SE_; ++j) mx = fmaxf(mx, sm[j]);
    red[0] = mx;
  }
  __syncthreads();
  float mx = red[0];
  if (tid < SE_) sm[tid] = expf(sm[tid] - mx);
  __syncthreads();
  if (tid == 0) {
    float s = 0.0f;
    for (int j = 0; j < SE_; ++j) s += sm[j];
    red[1] = s;
  }
  __syncthreads();
  if (tid < SE_)
    ctx[b * SE_ + tid] = (sm[tid] / red[1]) * upo_sum[b * SE_ + tid];
}

// ---------------- cross-entropy / loss ----------------

__global__ __launch_bounds__(256)
void ce_kernel(const float* __restrict__ logits, const int* __restrict__ d_tokens, int t,
               float* __restrict__ ce_masked, float* __restrict__ maskv) {
  __shared__ float red[256];
  int b = blockIdx.x, tid = threadIdx.x;
  const float* lg = logits + (size_t)b * DV_;
  float mx = -1e30f;
  for (int i = tid; i < DV_; i += 256) mx = fmaxf(mx, lg[i]);
  red[tid] = mx;
  __syncthreads();
  for (int s = 128; s > 0; s >>= 1) {
    if (tid < s) red[tid] = fmaxf(red[tid], red[tid + s]);
    __syncthreads();
  }
  mx = red[0];
  __syncthreads();
  float sum = 0.0f;
  for (int i = tid; i < DV_; i += 256) sum += expf(lg[i] - mx);
  red[tid] = sum;
  __syncthreads();
  for (int s = 128; s > 0; s >>= 1) {
    if (tid < s) red[tid] += red[tid + s];
    __syncthreads();
  }
  if (tid == 0) {
    float lse = mx + logf(red[0]);
    int lt = d_tokens[b * SD_ + t + 1];
    int lab = lt - 1;
    if (lab < 0) lab = 0;
    float ce = lse - lg[lab];
    ce_masked[b] = (lt != 0) ? ce : 0.0f;
    maskv[b]     = (lt != 0) ? 1.0f : 0.0f;
  }
}

__global__ void loss_step_kernel(const float* __restrict__ ce_masked,
                                 const float* __restrict__ maskv,
                                 float* __restrict__ loss_acc) {
  if (threadIdx.x == 0 && blockIdx.x == 0) {
    float cnt = 0.0f, s = 0.0f;
    for (int b = 0; b < B_; ++b) { cnt += maskv[b]; s += ce_masked[b]; }
    if (cnt > 0.0f) *loss_acc += s / cnt;
  }
}

__global__ void finalize_kernel(const float* __restrict__ loss_acc, float* __restrict__ out) {
  if (threadIdx.x == 0 && blockIdx.x == 0) out[0] = *loss_acc;
}

// ---------------- host orchestration ----------------

extern "C" void kernel_launch(void* const* d_in, const int* in_sizes, int n_in,
                              void* d_out, int out_size, void* d_ws, size_t ws_size,
                              hipStream_t stream) {
  (void)in_sizes; (void)n_in; (void)out_size; (void)ws_size;
  const int*   e_tokens  = (const int*)d_in[0];
  const int*   e_lengths = (const int*)d_in[1];
  const int*   d_tokens  = (const int*)d_in[2];
  const float* emb1_w    = (const float*)d_in[3];
  const float* emb2_w    = (const float*)d_in[4];
  const float* Wih1      = (const float*)d_in[5];
  const float* Whh1      = (const float*)d_in[6];
  const float* bih1      = (const float*)d_in[7];
  const float* bhh1      = (const float*)d_in[8];
  const float* W1        = (const float*)d_in[9];
  const float* b1        = (const float*)d_in[10];
  const float* W2        = (const float*)d_in[11];
  const float* b2        = (const float*)d_in[12];
  const float* Wih2      = (const float*)d_in[13];
  const float* Whh2      = (const float*)d_in[14];
  const float* bih2      = (const float*)d_in[15];
  const float* bhh2      = (const float*)d_in[16];
  const float* W3        = (const float*)d_in[17];
  const float* b3        = (const float*)d_in[18];

  // workspace carve (256B aligned)
  char* base = (char*)d_ws;
  size_t off = 0;
  auto carve = [&](size_t bytes) -> char* {
    off = (off + 255) & ~(size_t)255;
    char* p = base + off;
    off += bytes;
    return p;
  };
  const size_t nLW = (size_t)L_ * G4E * E_;            // 4.19M per stack
  __bf16* ws_Wih1 = (__bf16*)carve(nLW * 2);
  __bf16* ws_Whh1 = (__bf16*)carve(nLW * 2);
  __bf16* ws_Wih2 = (__bf16*)carve(nLW * 2);
  __bf16* ws_Whh2 = (__bf16*)carve(nLW * 2);
  __bf16* ws_W3   = (__bf16*)carve((size_t)DV_ * E_ * 2);
  __bf16* ws_W2p  = (__bf16*)carve((size_t)E_ * KPAD * 2);
  float*  ws_bs1  = (float*)carve((size_t)L_ * G4E * 4);
  float*  ws_bs2  = (float*)carve((size_t)L_ * G4E * 4);
  float*  ws_h    = (float*)carve((size_t)L_ * B_ * E_ * 4);
  float*  ws_c    = (float*)carve((size_t)L_ * B_ * E_ * 4);
  __bf16* ws_hbf  = (__bf16*)carve((size_t)L_ * B_ * E_ * 2);
  __bf16* ws_x0   = (__bf16*)carve((size_t)B_ * E_ * 2);
  __bf16* ws_dein = (__bf16*)carve((size_t)B_ * KPAD * 2);
  float*  ws_gates= (float*)carve((size_t)B_ * G4E * 4);
  float*  ws_upo  = (float*)carve((size_t)B_ * SE_ * E_ * 4);
  float*  ws_usum = (float*)carve((size_t)B_ * SE_ * 4);
  float*  ws_att  = (float*)carve((size_t)B_ * SE_ * 4);
  float*  ws_ctx  = (float*)carve((size_t)B_ * SE_ * 4);
  float*  ws_logit= (float*)carve((size_t)B_ * DV_ * 4);
  float*  ws_ce   = (float*)carve((size_t)B_ * 4);
  float*  ws_mask = (float*)carve((size_t)B_ * 4);
  float*  ws_loss = (float*)carve(256);

  const int BE = B_ * E_;           // 16384
  const int LBE = L_ * BE;          // 65536

  // ---- prep: bf16 weights, fused biases, zero state ----
  {
    int n = (int)nLW;
    int blk = (n + 255) / 256;
    cvt_bf16_kernel<<<blk, 256, 0, stream>>>(Wih1, ws_Wih1, n);
    cvt_bf16_kernel<<<blk, 256, 0, stream>>>(Whh1, ws_Whh1, n);
    cvt_bf16_kernel<<<blk, 256, 0, stream>>>(Wih2, ws_Wih2, n);
    cvt_bf16_kernel<<<blk, 256, 0, stream>>>(Whh2, ws_Whh2, n);
  }
  cvt_bf16_kernel<<<(DV_ * E_ + 255) / 256, 256, 0, stream>>>(W3, ws_W3, DV_ * E_);
  cvt_w2_pad_kernel<<<(E_ * KPAD + 255) / 256, 256, 0, stream>>>(W2, ws_W2p);
  bias_sum_kernel<<<(L_ * G4E + 255) / 256, 256, 0, stream>>>(bih1, bhh1, ws_bs1, L_ * G4E);
  bias_sum_kernel<<<(L_ * G4E + 255) / 256, 256, 0, stream>>>(bih2, bhh2, ws_bs2, L_ * G4E);
  zero_f32_kernel<<<(LBE + 255) / 256, 256, 0, stream>>>(ws_h, LBE);
  zero_f32_kernel<<<(LBE + 255) / 256, 256, 0, stream>>>(ws_c, LBE);
  zero_bf16_kernel<<<(LBE + 255) / 256, 256, 0, stream>>>(ws_hbf, LBE);
  zero_f32_kernel<<<1, 256, 0, stream>>>(ws_loss, 1);

  // ---- encoder: SE steps x L layers ----
  for (int t = 0; t < SE_; ++t) {
    gather_emb1_kernel<<<BE / 256, 256, 0, stream>>>(e_tokens, emb1_w, t, ws_x0);
    for (int l = 0; l < L_; ++l) {
      const __bf16* Ax = (l == 0) ? ws_x0 : (ws_hbf + (size_t)(l - 1) * BE);
      lstm_gates_kernel<<<16, 256, 0, stream>>>(
          Ax, ws_hbf + (size_t)l * BE,
          ws_Wih1 + (size_t)l * G4E * E_, ws_Whh1 + (size_t)l * G4E * E_,
          ws_bs1 + l * G4E, ws_gates);
      lstm_cell_kernel<<<BE / 256, 256, 0, stream>>>(
          ws_gates, ws_h + (size_t)l * BE, ws_c + (size_t)l * BE,
          ws_hbf + (size_t)l * BE, e_lengths, t, /*masked=*/1,
          ws_upo, (l == L_ - 1) ? 1 : 0);
    }
  }
  upo_sum_kernel<<<(B_ * SE_ + 255) / 256, 256, 0, stream>>>(ws_upo, ws_usum);

  // ---- decoder: SD-1 steps ----
  for (int t = 0; t < SD_ - 1; ++t) {
    att_dots_kernel<<<(B_ * SE_ * 32) / 256, 256, 0, stream>>>(
        ws_h + (size_t)3 * BE, ws_upo, ws_att);
    att_softmax_ctx_kernel<<<B_, 64, 0, stream>>>(ws_att, W1, b1, ws_usum, ws_ctx);
    build_de_in_kernel<<<(B_ * KPAD + 255) / 256, 256, 0, stream>>>(
        ws_ctx, emb2_w, d_tokens, t, ws_dein);
    // de = de_in @ W2^T + b2  -> bf16 x0   (N=512, K=576)
    wmma_gemm_kernel<true><<<(E_ / 16 + 7) / 8, 256, 0, stream>>>(
        ws_dein, ws_W2p, b2, (void*)ws_x0, E_, KPAD);
    for (int l = 0; l < L_; ++l) {
      const __bf16* Ax = (l == 0) ? ws_x0 : (ws_hbf + (size_t)(l - 1) * BE);
      lstm_gates_kernel<<<16, 256, 0, stream>>>(
          Ax, ws_hbf + (size_t)l * BE,
          ws_Wih2 + (size_t)l * G4E * E_, ws_Whh2 + (size_t)l * G4E * E_,
          ws_bs2 + l * G4E, ws_gates);
      lstm_cell_kernel<<<BE / 256, 256, 0, stream>>>(
          ws_gates, ws_h + (size_t)l * BE, ws_c + (size_t)l * BE,
          ws_hbf + (size_t)l * BE, e_lengths, t, /*masked=*/0,
          nullptr, 0);
    }
    // logits = h_top @ W3^T + b3   (N=16000, K=512)
    wmma_gemm_kernel<false><<<(DV_ / 16 + 7) / 8, 256, 0, stream>>>(
        ws_hbf + (size_t)3 * BE, ws_W3, b3, (void*)ws_logit, DV_, E_);
    ce_kernel<<<B_, 256, 0, stream>>>(ws_logit, d_tokens, t, ws_ce, ws_mask);
    loss_step_kernel<<<1, 32, 0, stream>>>(ws_ce, ws_mask, ws_loss);
  }

  finalize_kernel<<<1, 32, 0, stream>>>(ws_loss, (float*)d_out);
}
